// graphEncoder_58205396795604
// MI455X (gfx1250) — matrix operand order repaired and checked by reading, not compiled, not run on hardware
//
#include <hip/hip_runtime.h>
#include <hip/hip_bf16.h>

#define DIMC 128
#define NHEADS 8
#define NN 4096
#define ALPHA 0.2f
#define NEG_BIG (-9e15f)

typedef float    v2f  __attribute__((ext_vector_type(2)));
typedef float    v8f  __attribute__((ext_vector_type(8)));
typedef _Float16 v8h  __attribute__((ext_vector_type(8)));
typedef _Float16 v16h __attribute__((ext_vector_type(16)));

__device__ __forceinline__ float lrelu(float x) { return x > 0.f ? x : ALPHA * x; }
__device__ __forceinline__ float elu(float x)   { return x > 0.f ? x : __expf(x) - 1.f; }

// ---------------------------------------------------------------------------
// Kernel 1: Wh[h] = nodes @ W[h]  (f32 WMMA, full precision).
// Epilogue writes Wh (f32, row-major, for f1/f2/softmax) AND WhT (f16,
// transposed [h][d][node], packed b128 store per lane) for the attention
// B-operand fragments.
// ---------------------------------------------------------------------------
__global__ void wh_gemm_kernel(const float* __restrict__ nodes,
                               const float* __restrict__ W,
                               float* __restrict__ Wh,
                               _Float16* __restrict__ WhT) {
    const int h    = blockIdx.x >> 8;
    const int rb   = (blockIdx.x & 255) * 16;
    const int wv   = threadIdx.x >> 5;
    const int lane = threadIdx.x & 31;
    const int r    = lane & 15;
    const int koff = (lane >> 4) * 2;
    const int n    = wv * 16 + r;
    const float* __restrict__ Wb = W + h * DIMC * DIMC;

    v8f acc = {};
#pragma unroll 8
    for (int k0 = 0; k0 < DIMC; k0 += 4) {
        const int k = k0 + koff;
        v2f a, b;
        a.x = nodes[(rb + r) * DIMC + k];
        a.y = nodes[(rb + r) * DIMC + k + 1];
        b.x = Wb[k * DIMC + n];
        b.y = Wb[(k + 1) * DIMC + n];
        acc = __builtin_amdgcn_wmma_f32_16x16x4_f32(false, a, false, b,
                                                    (short)0, acc, false, false);
    }
    const int rbase = rb + ((lane < 16) ? 0 : 8);
    float* out = Wh + (size_t)h * NN * DIMC;
#pragma unroll
    for (int v = 0; v < 8; ++v)
        out[(rbase + v) * DIMC + n] = acc[v];

    // transposed f16 copy: WhT[h][n][node]; per-lane 8 consecutive k -> b128
    v8h hv;
#pragma unroll
    for (int v = 0; v < 8; ++v) hv[v] = (_Float16)acc[v];
    *(v8h*)(WhT + ((size_t)h * DIMC + n) * NN + rbase) = hv;
}

// ---------------------------------------------------------------------------
// Kernel 2: f1[h,n] = Wh[h,n,:].a1[h],  f2 likewise.  One wave per (h,n).
// ---------------------------------------------------------------------------
__global__ void f12_kernel(const float* __restrict__ Wh,
                           const float* __restrict__ a1,
                           const float* __restrict__ a2,
                           float* __restrict__ f1, float* __restrict__ f2) {
    const int gid  = blockIdx.x * 8 + (threadIdx.x >> 5);
    const int lane = threadIdx.x & 31;
    const int h    = gid >> 12;
    const int n    = gid & 4095;
    const float* row = Wh + ((size_t)h * NN + n) * DIMC;
    const float* A1  = a1 + h * DIMC;
    const float* A2  = a2 + h * DIMC;
    float s1 = 0.f, s2 = 0.f;
#pragma unroll
    for (int i = lane; i < DIMC; i += 32) {
        const float w = row[i];
        s1 += w * A1[i];
        s2 += w * A2[i];
    }
    for (int off = 16; off; off >>= 1) {
        s1 += __shfl_down(s1, off, 32);
        s2 += __shfl_down(s2, off, 32);
    }
    if (lane == 0) { f1[gid] = s1; f2[gid] = s2; }
}

// ---------------------------------------------------------------------------
// Kernel 3: per-row online softmax stats over masked leaky-relu scores.
// One wave per (h, row): m = rowmax, linv = 1/rowsum(exp(e-m)).
// ---------------------------------------------------------------------------
__global__ void stats_kernel(const int* __restrict__ adj,
                             const float* __restrict__ f1,
                             const float* __restrict__ f2,
                             float* __restrict__ mrow, float* __restrict__ linv) {
    const int gid  = blockIdx.x * 8 + (threadIdx.x >> 5);
    const int lane = threadIdx.x & 31;
    const int h    = gid >> 12;
    const int n    = gid & 4095;
    const float f1r = f1[gid];
    const float* __restrict__ F2 = f2 + (h << 12);
    const int*   __restrict__ ar = adj + (size_t)n * NN;

    float m = NEG_BIG, l = 0.f;
    for (int c = lane; c < NN; c += 32) {
        const float e = (ar[c] > 0) ? lrelu(f1r + F2[c]) : NEG_BIG;
        if (e > m) { l *= __expf(m - e); m = e; }
        l += __expf(e - m);
    }
    for (int off = 16; off; off >>= 1) {
        const float m2 = __shfl_down(m, off, 32);
        const float l2 = __shfl_down(l, off, 32);
        const float mm = fmaxf(m, m2);
        l = l * __expf(m - mm) + l2 * __expf(m2 - mm);
        m = mm;
    }
    if (lane == 0) { mrow[gid] = m; linv[gid] = 1.f / l; }
}

// ---------------------------------------------------------------------------
// Kernel 4: h[h,rows,:] = (exp(e-m)/l) @ Wh[h]  via f16 WMMA 16x16x32,
// f32 accumulate (flash-attention precision scheme).  Block = (head, 16
// rows), 8 waves = 8 D-tiles.  p staged in f16 LDS (row stride 264 halves:
// 528 mod 256 == 16 -> conflict-free A-fragment reads).  B operand comes
// from transposed f16 WhT -> one contiguous 32B read per lane per WMMA.
// ---------------------------------------------------------------------------
__global__ void attn_kernel(const int* __restrict__ adj,
                            const float* __restrict__ f1,
                            const float* __restrict__ f2,
                            const float* __restrict__ mrow,
                            const float* __restrict__ linv,
                            const _Float16* __restrict__ WhT,
                            float* __restrict__ hcat) {
    __shared__ _Float16 p_lds[16][264];

    const int h    = blockIdx.x >> 8;
    const int rb   = (blockIdx.x & 255) * 16;
    const int t    = threadIdx.x;
    const int wv   = t >> 5;
    const int lane = t & 31;

    // p-generation role: 16 threads per row, 16 cols each (strided)
    const int pr = t >> 4;
    const int pj = t & 15;
    const int hbase = (h << 12);
    const float frow = f1[hbase + rb + pr];
    const float mr   = mrow[hbase + rb + pr];
    const float lr   = linv[hbase + rb + pr];
    const int*   __restrict__ arow = adj + (size_t)(rb + pr) * NN;
    const float* __restrict__ F2   = f2 + hbase;

    // WMMA roles
    const int rA    = lane & 15;
    const int ko    = (lane < 16) ? 0 : 16;   // A-frag K offset (contiguous k = 16h + s, same as B)
    const int koffB = (lane < 16) ? 0 : 16;   // B-frag K offset
    const int nB    = wv * 16 + rA;
    const _Float16* __restrict__ bcol =
        WhT + ((size_t)h * DIMC + nB) * NN + koffB;

    v8f acc = {};
    for (int cb = 0; cb < NN; cb += 256) {
        // normalized attention weights for this 16x256 tile -> f16 LDS
#pragma unroll
        for (int i = 0; i < 16; ++i) {
            const int cl = pj + 16 * i;
            const int c  = cb + cl;
            const float e = (arow[c] > 0) ? lrelu(frow + F2[c]) : NEG_BIG;
            p_lds[pr][cl] = (_Float16)(__expf(e - mr) * lr);
        }
        __syncthreads();
        // acc += p_tile @ Wh_tile   (8 f16 WMMAs per wave, K=32 each)
#pragma unroll
        for (int k0 = 0; k0 < 256; k0 += 32) {
            const _Float16* ap = &p_lds[rA][k0 + ko];
            const v8h alo = *(const v8h*)(ap);
            const v8h ahi = *(const v8h*)(ap + 8);
            const v16h a  = __builtin_shufflevector(alo, ahi,
                              0,1,2,3,4,5,6,7,8,9,10,11,12,13,14,15);
            const v16h b  = *(const v16h*)(bcol + cb + k0);
            acc = __builtin_amdgcn_wmma_f32_16x16x32_f16(false, a, false, b,
                                                         (short)0, acc, false, false);
        }
        __syncthreads();
    }

    // ELU + scatter to concat layout [N, H*D]
    const int rbase = rb + ((lane < 16) ? 0 : 8);
#pragma unroll
    for (int v = 0; v < 8; ++v)
        hcat[(size_t)(rbase + v) * (NHEADS * DIMC) + h * DIMC + nB] = elu(acc[v]);
}

// ---------------------------------------------------------------------------
// Kernel 5: x = elu(hcat @ W1 + b1); out = LayerNorm(nodes + x).
// f32 WMMA (only 1 GFLOP).  Block = 16 rows; 8 waves = 8 D-tiles.
// ---------------------------------------------------------------------------
__global__ void mlp_ln_kernel(const float* __restrict__ hcat,
                              const float* __restrict__ W1,
                              const float* __restrict__ b1,
                              const float* __restrict__ nodes,
                              const float* __restrict__ gamma,
                              const float* __restrict__ beta,
                              float* __restrict__ out) {
    __shared__ float y_lds[16][129];

    const int rb   = blockIdx.x * 16;
    const int t    = threadIdx.x;
    const int wv   = t >> 5;
    const int lane = t & 31;
    const int rA   = lane & 15;
    const int koff = (lane >> 4) * 2;
    const int nB   = wv * 16 + rA;

    v8f acc = {};
#pragma unroll 4
    for (int k0 = 0; k0 < NHEADS * DIMC; k0 += 4) {
        const int k = k0 + koff;
        v2f a, b;
        a.x = hcat[(size_t)(rb + rA) * (NHEADS * DIMC) + k];
        a.y = hcat[(size_t)(rb + rA) * (NHEADS * DIMC) + k + 1];
        b.x = W1[k * DIMC + nB];
        b.y = W1[(k + 1) * DIMC + nB];
        acc = __builtin_amdgcn_wmma_f32_16x16x4_f32(false, a, false, b,
                                                    (short)0, acc, false, false);
    }

    const int mh = (lane < 16) ? 0 : 8;
    const float bb = b1[nB];
#pragma unroll
    for (int v = 0; v < 8; ++v) {
        const int M = v + mh;
        const float x = elu(acc[v] + bb);
        y_lds[M][nB] = nodes[(rb + M) * DIMC + nB] + x;
    }
    __syncthreads();

    const int r = t >> 4, j = t & 15;
    float s = 0.f, ss = 0.f;
#pragma unroll
    for (int i = 0; i < 8; ++i) {
        const float y = y_lds[r][j + 16 * i];
        s += y; ss += y * y;
    }
    for (int off = 8; off; off >>= 1) {
        s  += __shfl_xor(s,  off, 16);
        ss += __shfl_xor(ss, off, 16);
    }
    const float mu   = s * (1.f / DIMC);
    const float var  = ss * (1.f / DIMC) - mu * mu;
    const float rstd = rsqrtf(var + 1e-5f);
#pragma unroll
    for (int i = 0; i < 8; ++i) {
        const int c = j + 16 * i;
        const float y = y_lds[r][c];
        out[(rb + r) * DIMC + c] = (y - mu) * rstd * gamma[c] + beta[c];
    }
}

// ---------------------------------------------------------------------------
extern "C" void kernel_launch(void* const* d_in, const int* in_sizes, int n_in,
                              void* d_out, int out_size, void* d_ws, size_t ws_size,
                              hipStream_t stream) {
    const float* nodes = (const float*)d_in[0];
    const int*   adj   = (const int*)  d_in[1];
    const float* W     = (const float*)d_in[2];
    const float* a1    = (const float*)d_in[3];
    const float* a2    = (const float*)d_in[4];
    const float* W1    = (const float*)d_in[5];
    const float* b1    = (const float*)d_in[6];
    const float* gamma = (const float*)d_in[7];
    const float* beta  = (const float*)d_in[8];
    float* out = (float*)d_out;

    // workspace: Wh f32 16MB | f1/f2/m/linv | hcat f32 16MB | WhT f16 8MB
    float* Wh   = (float*)d_ws;
    float* f1   = Wh   + (size_t)NHEADS * NN * DIMC;
    float* f2   = f1   + (size_t)NHEADS * NN;
    float* mrow = f2   + (size_t)NHEADS * NN;
    float* linv = mrow + (size_t)NHEADS * NN;
    float* hcat = linv + (size_t)NHEADS * NN;
    _Float16* WhT = (_Float16*)(hcat + (size_t)NN * NHEADS * DIMC);

    wh_gemm_kernel<<<dim3(NHEADS * (NN / 16)), dim3(256), 0, stream>>>(nodes, W, Wh, WhT);
    f12_kernel    <<<dim3(NHEADS * NN / 8),    dim3(256), 0, stream>>>(Wh, a1, a2, f1, f2);
    stats_kernel  <<<dim3(NHEADS * NN / 8),    dim3(256), 0, stream>>>(adj, f1, f2, mrow, linv);
    attn_kernel   <<<dim3(NHEADS * (NN / 16)), dim3(256), 0, stream>>>(adj, f1, f2, mrow, linv, WhT, hcat);
    mlp_ln_kernel <<<dim3(NN / 16),            dim3(256), 0, stream>>>(hcat, W1, b1, nodes, gamma, beta, out);
}
